// Mamba_27960237097102
// MI455X (gfx1250) — matrix operand order repaired and checked by reading, not compiled, not run on hardware
//
#include <hip/hip_runtime.h>
#include <hip/hip_bf16.h>
#include <stdint.h>

// Model dims (match reference)
#define D_MODEL 1024
#define VOCAB   50264
#define SEQ     2048
#define NLAYER  4
#define DSTATE  16
#define DINNER  2048
#define DTRANK  64
#define KCONV   4
#define EPSNORM 1e-6f
#define XDBLW   (DTRANK + 2 * DSTATE)   // 96

typedef __bf16 bf16_t;
typedef bf16_t v16bf  __attribute__((ext_vector_type(16)));
typedef bf16_t bf16x2 __attribute__((ext_vector_type(2)));
typedef float  v8f    __attribute__((ext_vector_type(8)));

union FragBF { v16bf v; uint4 q[2]; };

// Native bf16 conversion: let clang pick v_cvt_pk_bf16_f32-class codegen.
__device__ __forceinline__ void st_bf2(unsigned short* p, float lo, float hi) {
  bf16x2 v;
  v.x = (bf16_t)lo;
  v.y = (bf16_t)hi;
  *(bf16x2*)p = v;
}
__device__ __forceinline__ float siluf(float x) { return x / (1.f + __expf(-x)); }
__device__ __forceinline__ float softplusf(float x) {
  return (x > 20.f) ? x : log1pf(__expf(x));
}

// ---------------------------------------------------------------------------
// C[M,N] = act(A[M,K](lda) @ B[N,K]^T + bias) (+ addsrc), bf16 WMMA, f32 acc.
// Block: 256 threads (8 waves), tile 128x128, K-step 32.
// Wave tile: 32(M) x 64(N) = 2x4 v_wmma_f32_16x16x32_bf16 accumulators.
// Grid: blockIdx.x = M-tile (fast) so concurrent blocks share B tiles in L2.
// ---------------------------------------------------------------------------
#define BM 128
#define BN 128
#define BK 32
#define LDSA 40   // padded LDS row stride (bf16 elems): conflict-free frag reads

__global__ __launch_bounds__(256) void gemm_bf16_wmma(
    const float* __restrict__ A, int lda,
    const float* __restrict__ B,                 // [N,K], ld = K
    float* __restrict__ C,                       // [M,N]
    int M, int N, int K,
    const float* __restrict__ bias,              // per-column, may be null
    const float* __restrict__ addsrc,            // [M,N] residual, may be null
    int act)                                     // 0=none, 1=softplus
{
  __shared__ unsigned short sA[BM * LDSA];
  __shared__ unsigned short sB[BN * LDSA];

  const int tid  = threadIdx.x;
  const int lane = tid & 31;
  const int wave = tid >> 5;
  const int wm   = wave & 3;        // 4 M-groups of 32 rows
  const int wn   = wave >> 2;       // 2 N-groups of 64 cols
  const int half = lane >> 4;
  const int idx  = lane & 15;

  const int tileM = blockIdx.x * BM;   // M fastest-varying: B-tile L2 reuse
  const int tileN = blockIdx.y * BN;

  // cooperative staging: 128 rows x 32 cols for A and for B; 16 floats/thread
  const int sRow = tid >> 1;
  const int sCol = (tid & 1) * 16;
  const bool bValid = (tileN + sRow) < N;

  const float* aG = A + (size_t)(tileM + sRow) * lda + sCol;
  const float* bG = bValid ? (B + (size_t)(tileN + sRow) * K + sCol) : A;

  v8f acc[2][4] = {};

  for (int kb = 0; kb < K; kb += BK) {
    {   // stage A tile (f32 -> bf16)
      const float4* p = (const float4*)(aG + kb);
      float4 f0 = p[0], f1 = p[1], f2 = p[2], f3 = p[3];
      unsigned short* s = &sA[sRow * LDSA + sCol];
      st_bf2(s + 0,  f0.x, f0.y); st_bf2(s + 2,  f0.z, f0.w);
      st_bf2(s + 4,  f1.x, f1.y); st_bf2(s + 6,  f1.z, f1.w);
      st_bf2(s + 8,  f2.x, f2.y); st_bf2(s + 10, f2.z, f2.w);
      st_bf2(s + 12, f3.x, f3.y); st_bf2(s + 14, f3.z, f3.w);
    }
    {   // stage B tile (zero-fill OOB rows of W)
      unsigned short* s = &sB[sRow * LDSA + sCol];
      if (bValid) {
        const float4* p = (const float4*)(bG + kb);
        float4 f0 = p[0], f1 = p[1], f2 = p[2], f3 = p[3];
        st_bf2(s + 0,  f0.x, f0.y); st_bf2(s + 2,  f0.z, f0.w);
        st_bf2(s + 4,  f1.x, f1.y); st_bf2(s + 6,  f1.z, f1.w);
        st_bf2(s + 8,  f2.x, f2.y); st_bf2(s + 10, f2.z, f2.w);
        st_bf2(s + 12, f3.x, f3.y); st_bf2(s + 14, f3.z, f3.w);
      } else {
        unsigned int* si = (unsigned int*)s;
        si[0] = 0u; si[1] = 0u; si[2] = 0u; si[3] = 0u;
        si[4] = 0u; si[5] = 0u; si[6] = 0u; si[7] = 0u;
      }
    }
    __syncthreads();

    if (kb + BK < K) {            // gfx1250 global_prefetch_b8 on next K tile
      __builtin_prefetch(aG + kb + BK, 0, 0);
      if (bValid) __builtin_prefetch(bG + kb + BK, 0, 0);
    }

#pragma unroll
    for (int mi = 0; mi < 2; ++mi) {
      FragBF fa;   // A layout: half0 -> K{0..7,16..23}, half1 -> K{8..15,24..31}
      const int rA = (wm * 32 + mi * 16 + idx) * LDSA;
      fa.q[0] = *(const uint4*)&sA[rA + half * 8];
      fa.q[1] = *(const uint4*)&sA[rA + 16 + half * 8];
#pragma unroll
      for (int ni = 0; ni < 4; ++ni) {
        FragBF fb; // B layout: lane n; half0 -> K{0..7,16..23}, half1 -> K{8..15,24..31}
        const int rB = (wn * 64 + ni * 16 + idx) * LDSA + half * 8;
        fb.q[0] = *(const uint4*)&sB[rB];
        fb.q[1] = *(const uint4*)&sB[rB + 16];
        acc[mi][ni] = __builtin_amdgcn_wmma_f32_16x16x32_bf16(
            false, fa.v, false, fb.v, (short)0, acc[mi][ni], false, false);
      }
    }
    __syncthreads();
  }

  // epilogue: D layout — VGPR r holds (row = r + 8*half, col = idx)
#pragma unroll
  for (int mi = 0; mi < 2; ++mi) {
#pragma unroll
    for (int ni = 0; ni < 4; ++ni) {
      const int col = tileN + wn * 64 + ni * 16 + idx;
      if (col >= N) continue;
      const int rowBase = tileM + wm * 32 + mi * 16 + half * 8;
      const float bv = bias ? bias[col] : 0.f;
#pragma unroll
      for (int r = 0; r < 8; ++r) {
        float v = acc[mi][ni][r] + bv;
        if (act == 1) v = softplusf(v);
        const size_t off = (size_t)(rowBase + r) * N + col;
        if (addsrc) v += addsrc[off];
        C[off] = v;
      }
    }
  }
}

// ---------------------------------------------------------------------------
__global__ __launch_bounds__(256) void embed_kernel(
    const int* __restrict__ tokens, const float* __restrict__ emb,
    float* __restrict__ x) {
  const int t = blockIdx.x;
  const int tok = tokens[t];
  const float* src = emb + (size_t)tok * D_MODEL;
  float* dst = x + (size_t)t * D_MODEL;
  for (int i = threadIdx.x; i < D_MODEL; i += 256) dst[i] = src[i];
}

__global__ __launch_bounds__(256) void rmsnorm_kernel(
    const float* __restrict__ x, const float* __restrict__ w,
    const float* __restrict__ b, float* __restrict__ out) {
  __shared__ float red[256];
  const int row = blockIdx.x;
  const float* xr = x + (size_t)row * D_MODEL;
  float s = 0.f;
  for (int i = threadIdx.x; i < D_MODEL; i += 256) { float v = xr[i]; s += v * v; }
  red[threadIdx.x] = s; __syncthreads();
  for (int o = 128; o > 0; o >>= 1) {
    if (threadIdx.x < o) red[threadIdx.x] += red[threadIdx.x + o];
    __syncthreads();
  }
  const float inv = rsqrtf(red[0] / (float)D_MODEL + EPSNORM);
  float* dst = out + (size_t)row * D_MODEL;
  for (int i = threadIdx.x; i < D_MODEL; i += 256) dst[i] = w[i] * (xr[i] * inv) + b[i];
}

// depthwise causal conv (K=4) + bias + SiLU; input = xr[:, :DI] (stride 2*DI)
__global__ __launch_bounds__(256) void conv_silu_kernel(
    const float* __restrict__ xr, const float* __restrict__ w,
    const float* __restrict__ b, float* __restrict__ xs) {
  const int g = blockIdx.x * 256 + threadIdx.x;
  const int d = g & (DINNER - 1);
  const int t = g >> 11;                       // DINNER == 2048
  float acc = b[d];
#pragma unroll
  for (int k = 0; k < KCONV; ++k) {
    const int tt = t + k - (KCONV - 1);
    if (tt >= 0) acc += w[d * KCONV + k] * xr[(size_t)tt * (2 * DINNER) + d];
  }
  xs[g] = siluf(acc);
}

// selective scan: one lane per channel, h[16] in registers, sequential in t.
// Fused: + u*Dvec and * silu(res) gate.
__global__ __launch_bounds__(256) void scan_kernel(
    const float* __restrict__ xs, const float* __restrict__ delta,
    const float* __restrict__ xdbl, const float* __restrict__ xr,
    const float* __restrict__ A_log, const float* __restrict__ Dp,
    float* __restrict__ y) {
  const int d = blockIdx.x * 256 + threadIdx.x;     // 0..DINNER-1
  float Av[DSTATE], h[DSTATE];
#pragma unroll
  for (int s = 0; s < DSTATE; ++s) {
    Av[s] = -__expf(A_log[d * DSTATE + s]);
    h[s] = 0.f;
  }
  const float Dv = Dp[d];
  for (int t = 0; t < SEQ; ++t) {
    const float dl = delta[(size_t)t * DINNER + d];
    const float u  = xs[(size_t)t * DINNER + d];
    const float du = dl * u;
    const float* bc = xdbl + (size_t)t * XDBLW + DTRANK;   // B then C, 16 each
    float yv = 0.f;
#pragma unroll
    for (int s = 0; s < DSTATE; ++s) {
      h[s] = __expf(dl * Av[s]) * h[s] + du * bc[s];
      yv += h[s] * bc[DSTATE + s];
    }
    const float res = xr[(size_t)t * (2 * DINNER) + DINNER + d];
    y[(size_t)t * DINNER + d] = (yv + u * Dv) * siluf(res);
  }
}

// ---------------------------------------------------------------------------
extern "C" void kernel_launch(void* const* d_in, const int* in_sizes, int n_in,
                              void* d_out, int out_size, void* d_ws, size_t ws_size,
                              hipStream_t stream) {
  const int*   tokens  = (const int*)  d_in[0];
  const float* emb_W   = (const float*)d_in[1];
  const float* rms_w   = (const float*)d_in[2];
  const float* rms_b   = (const float*)d_in[3];
  const float* in_proj = (const float*)d_in[4];
  const float* conv_W  = (const float*)d_in[5];
  const float* conv_b  = (const float*)d_in[6];
  const float* x_proj  = (const float*)d_in[7];
  const float* dt_W    = (const float*)d_in[8];
  const float* dt_b    = (const float*)d_in[9];
  const float* A_log   = (const float*)d_in[10];
  const float* Dp      = (const float*)d_in[11];
  const float* out_W   = (const float*)d_in[12];
  const float* normf_w = (const float*)d_in[13];
  const float* normf_b = (const float*)d_in[14];
  float* out = (float*)d_out;

  float* ws   = (float*)d_ws;
  float* x    = ws;                                          // [L,D]
  float* xn   = x    + (size_t)SEQ * D_MODEL;                // [L,D]
  float* xr   = xn   + (size_t)SEQ * D_MODEL;                // [L,2*DI]
  float* xs   = xr   + (size_t)SEQ * 2 * DINNER;             // [L,DI]
  float* xdbl = xs   + (size_t)SEQ * DINNER;                 // [L,96]
  float* dlt  = xdbl + (size_t)SEQ * XDBLW;                  // [L,DI]
  float* yb   = dlt  + (size_t)SEQ * DINNER;                 // [L,DI]

  const dim3 blk(256);
  const int mTiles = SEQ / BM;                               // 16
  embed_kernel<<<SEQ, blk, 0, stream>>>(tokens, emb_W, x);

  for (int i = 0; i < NLAYER; ++i) {
    rmsnorm_kernel<<<SEQ, blk, 0, stream>>>(x, rms_w + i * D_MODEL,
                                            rms_b + i * D_MODEL, xn);
    // xr = xn @ in_proj^T                               [2048 x 4096 x 1024]
    gemm_bf16_wmma<<<dim3(mTiles, (2 * DINNER) / BN), blk, 0, stream>>>(
        xn, D_MODEL, in_proj + (size_t)i * 2 * DINNER * D_MODEL, xr,
        SEQ, 2 * DINNER, D_MODEL, nullptr, nullptr, 0);
    conv_silu_kernel<<<(SEQ * DINNER) / 256, blk, 0, stream>>>(
        xr, conv_W + (size_t)i * DINNER * KCONV, conv_b + (size_t)i * DINNER, xs);
    // xdbl = xs @ x_proj^T                              [2048 x 96 x 2048]
    gemm_bf16_wmma<<<dim3(mTiles, (XDBLW + BN - 1) / BN), blk, 0, stream>>>(
        xs, DINNER, x_proj + (size_t)i * XDBLW * DINNER, xdbl,
        SEQ, XDBLW, DINNER, nullptr, nullptr, 0);
    // delta = softplus(xdbl[:, :64] @ dt_W^T + dt_b)    [2048 x 2048 x 64]
    gemm_bf16_wmma<<<dim3(mTiles, DINNER / BN), blk, 0, stream>>>(
        xdbl, XDBLW, dt_W + (size_t)i * DINNER * DTRANK, dlt,
        SEQ, DINNER, DTRANK, dt_b + (size_t)i * DINNER, nullptr, 1);
    scan_kernel<<<DINNER / 256, blk, 0, stream>>>(
        xs, dlt, xdbl, xr, A_log + (size_t)i * DINNER * DSTATE,
        Dp + (size_t)i * DINNER, yb);
    // x = yb @ out_W^T + x (residual)                   [2048 x 1024 x 2048]
    gemm_bf16_wmma<<<dim3(mTiles, D_MODEL / BN), blk, 0, stream>>>(
        yb, DINNER, out_W + (size_t)i * D_MODEL * DINNER, x,
        SEQ, D_MODEL, DINNER, nullptr, x, 0);
  }

  rmsnorm_kernel<<<SEQ, blk, 0, stream>>>(x, normf_w, normf_b, xn);
  // logits = xn @ emb_W^T                               [2048 x 50264 x 1024]
  gemm_bf16_wmma<<<dim3(mTiles, (VOCAB + BN - 1) / BN), blk, 0, stream>>>(
      xn, D_MODEL, emb_W, out, SEQ, VOCAB, D_MODEL, nullptr, nullptr, 0);
}
